// DeepBatchModel_17300128269008
// MI455X (gfx1250) — matrix operand order repaired and checked by reading, not compiled, not run on hardware
//
#include <hip/hip_runtime.h>
#include <hip/hip_bf16.h>

#define NGRAPH 200
#define NPG    100
#define NNODES (NGRAPH * NPG)     // 20000
#define EPG    800
#define NEDGES (NGRAPH * EPG)     // 160000
#define IN_DIM 21
#define HID    256
#define NLAYER 4
#define EPS_BN 1e-5f
#define SLOPE  0.2f

typedef __attribute__((ext_vector_type(16))) _Float16 v16h;
typedef __attribute__((ext_vector_type(8)))  _Float16 v8h;
typedef __attribute__((ext_vector_type(4)))  _Float16 v4h;
typedef __attribute__((ext_vector_type(8)))  float    v8f;

// ---------------------------------------------------------------------------
// Generic GEMM: C[M,NC] = A[M,K] @ B[K,NC] (+bias), fp32 in/out, f16 WMMA math.
// Tile 64x64, 128 threads (4 waves), K stepped by 32 through LDS.
// Fragment layouts follow CDNA5 ISA 7.12.2 (16-bit A 16x32, B 32x16, f32 C 16x16).
// All fragments are loaded before the WMMA chain so the ds_load_b128s batch
// under one s_wait_dscnt and the four v_wmma issue back-to-back.
// ---------------------------------------------------------------------------
__global__ __launch_bounds__(128)
void gemm_wmma_kernel(const float* __restrict__ A, const float* __restrict__ B,
                      const float* __restrict__ bias, float* __restrict__ C,
                      int M, int K, int NC)
{
    __shared__ __align__(16) _Float16 As[64 * 32];   // [row][k]
    __shared__ __align__(16) _Float16 Bs[64 * 32];   // [col][k]

    const int tid  = threadIdx.x;
    const int wave = tid >> 5;
    const int lane = tid & 31;
    const int half = lane >> 4;
    const int lr   = lane & 15;

    const int m0 = blockIdx.x * 64;
    const int n0 = blockIdx.y * 64;

    const bool k_vec_ok = ((K & 3) == 0);

    v8f acc[4] = {};

    for (int k0 = 0; k0 < K; k0 += 32) {
        // Prefetch next K-tile into the WGP-near cache while this one runs.
        if (k0 + 32 < K) {
            int pr = m0 + (tid >> 1);
            pr = pr < M ? pr : M - 1;
            __builtin_prefetch(&A[(size_t)pr * K + k0 + 32], 0, 3);
            int pk = k0 + 32 + (tid & 31);
            pk = pk < K ? pk : K - 1;
            __builtin_prefetch(&B[(size_t)pk * NC + n0 + ((tid >> 5) << 4)], 0, 3);
        }

        const bool fullA = k_vec_ok && (m0 + 64 <= M) && (k0 + 32 <= K);
        const bool fullB = k_vec_ok && (k0 + 32 <= K);

        // ---- Stage A tile 64x32 as f16 ----
        if (fullA) {
            // 512 float4 groups; thread t handles groups t, t+128, ...
            for (int i = 0; i < 4; ++i) {
                int idx4 = tid + i * 128;
                int r  = idx4 >> 3;          // 8 float4 per 32-wide row
                int c4 = (idx4 & 7) * 4;
                const float4 v = *(const float4*)&A[(size_t)(m0 + r) * K + k0 + c4];
                v4h p = { (_Float16)v.x, (_Float16)v.y, (_Float16)v.z, (_Float16)v.w };
                *(v4h*)&As[r * 32 + c4] = p;
            }
        } else {
            // Branchless scalar path: clamp address, select-zero out-of-range.
            for (int t = tid; t < 64 * 32; t += 128) {
                int r = t >> 5, kk = t & 31;
                int gr = m0 + r, gk = k0 + kk;
                int cr = gr < M ? gr : M - 1;
                int ck = gk < K ? gk : K - 1;
                float v = A[(size_t)cr * K + ck];
                As[t] = (_Float16)((gr < M && gk < K) ? v : 0.0f);
            }
        }

        // ---- Stage B tile 32x64 as f16, transposed to [col][k] ----
        if (fullB) {
            for (int i = 0; i < 4; ++i) {
                int idx4 = tid + i * 128;
                int kk = idx4 >> 4;          // 16 float4 per 64-wide row
                int c4 = (idx4 & 15) * 4;
                const float4 v = *(const float4*)&B[(size_t)(k0 + kk) * NC + n0 + c4];
                Bs[(c4 + 0) * 32 + kk] = (_Float16)v.x;
                Bs[(c4 + 1) * 32 + kk] = (_Float16)v.y;
                Bs[(c4 + 2) * 32 + kk] = (_Float16)v.z;
                Bs[(c4 + 3) * 32 + kk] = (_Float16)v.w;
            }
        } else {
            for (int t = tid; t < 64 * 32; t += 128) {
                int n = t & 63, kk = t >> 6;
                int gk = k0 + kk;
                int ck = gk < K ? gk : K - 1;
                float v = B[(size_t)ck * NC + (n0 + n)];
                Bs[n * 32 + kk] = (_Float16)(gk < K ? v : 0.0f);
            }
        }
        __syncthreads();

        // A fragment: lane row = lr, K runs {half*8..+7} then {16+half*8..+7}.
        const _Float16* arow = &As[(wave * 16 + lr) * 32 + half * 8];
        v8h alo = *(const v8h*)(arow);
        v8h ahi = *(const v8h*)(arow + 16);
        v16h afrag = __builtin_shufflevector(alo, ahi,
            0, 1, 2, 3, 4, 5, 6, 7, 8, 9, 10, 11, 12, 13, 14, 15);

        // Load all four B fragments first so the ds_loads batch and the
        // four WMMAs can issue back-to-back.
        v16h bfrag[4];
        for (int j = 0; j < 4; ++j) {
            const _Float16* brow = &Bs[(j * 16 + lr) * 32 + half * 8];
            v8h blo = *(const v8h*)(brow);
            v8h bhi = *(const v8h*)(brow + 16);
            bfrag[j] = __builtin_shufflevector(blo, bhi,
                0, 1, 2, 3, 4, 5, 6, 7, 8, 9, 10, 11, 12, 13, 14, 15);
        }
        for (int j = 0; j < 4; ++j) {
            acc[j] = __builtin_amdgcn_wmma_f32_16x16x32_f16(
                false, afrag, false, bfrag[j], (short)0, acc[j], false, false);
        }
        __syncthreads();
    }

    // C layout: VGPR r -> row (wave*16 + half*8 + r), col = n0 + j*16 + lr.
    for (int j = 0; j < 4; ++j) {
        int col = n0 + j * 16 + lr;
        float bv = bias ? bias[col] : 0.0f;
        for (int r = 0; r < 8; ++r) {
            int row = m0 + wave * 16 + half * 8 + r;
            if (row < M && col < NC) C[(size_t)row * NC + col] = acc[j][r] + bv;
        }
    }
}

// ---------------------------------------------------------------------------
// Utility fills / copies
// ---------------------------------------------------------------------------
__global__ void fill_f32(float* p, float v, int n) {
    int i = blockIdx.x * blockDim.x + threadIdx.x;
    if (i < n) p[i] = v;
}
__global__ void fill_u32(unsigned* p, unsigned v, int n) {
    int i = blockIdx.x * blockDim.x + threadIdx.x;
    if (i < n) p[i] = v;
}
__global__ void copy_f32(float* __restrict__ d, const float* __restrict__ s, int n) {
    int i = blockIdx.x * blockDim.x + threadIdx.x;
    if (i < n) d[i] = s[i];
}

// ---------------------------------------------------------------------------
// Edge score: one wave32 per edge; segment-max via ordered-uint atomicMax.
// ---------------------------------------------------------------------------
__device__ __forceinline__ unsigned enc_f32(float f) {
    unsigned u = __float_as_uint(f);
    return (u & 0x80000000u) ? ~u : (u | 0x80000000u);
}
__device__ __forceinline__ float dec_f32(unsigned u) {
    return (u & 0x80000000u) ? __uint_as_float(u & 0x7fffffffu)
                             : __uint_as_float(~u);
}

__global__ __launch_bounds__(256)
void edge_score_kernel(const float* __restrict__ fs, const float* __restrict__ fd,
                       const float* __restrict__ av,
                       const int* __restrict__ src, const int* __restrict__ dst,
                       float* __restrict__ score, unsigned* __restrict__ nodemax, int E)
{
    int e = blockIdx.x * 8 + (threadIdx.x >> 5);
    if (e >= E) return;
    int lane = threadIdx.x & 31;
    const float* ps = fs + (size_t)src[e] * HID;
    const float* pd = fd + (size_t)dst[e] * HID;
    float sum = 0.0f;
    for (int i = lane; i < HID; i += 32) {
        float v = ps[i] + pd[i];
        v = v > 0.0f ? v : SLOPE * v;    // GATv2 leaky_relu
        sum += v * av[i];
    }
    for (int off = 16; off > 0; off >>= 1)
        sum += __shfl_down(sum, off, 32);
    if (lane == 0) {
        score[e] = sum;
        atomicMax(nodemax + dst[e], enc_f32(sum));
    }
}

__global__ void edge_exp_kernel(const float* __restrict__ score,
                                const int* __restrict__ dst,
                                const unsigned* __restrict__ nodemax,
                                float* __restrict__ ex, float* __restrict__ den, int E)
{
    int e = blockIdx.x * blockDim.x + threadIdx.x;
    if (e >= E) return;
    int d = dst[e];
    float v = expf(score[e] - dec_f32(nodemax[d]));
    ex[e] = v;
    atomicAdd(den + d, v);
}

__global__ void alpha_kernel(const float* __restrict__ ex, const float* __restrict__ den,
                             const int* __restrict__ dst,
                             float* __restrict__ attn_out, int layer, int E)
{
    int e = blockIdx.x * blockDim.x + threadIdx.x;
    if (e >= E) return;
    attn_out[(size_t)e * NLAYER + layer] = ex[e] / den[dst[e]];
}

// rst[dst] += alpha * fs[src], one thread per (edge, channel)
__global__ void agg_kernel(const float* __restrict__ fs, const float* __restrict__ attn,
                           const int* __restrict__ src, const int* __restrict__ dst,
                           float* __restrict__ rst, int layer)
{
    int idx = blockIdx.x * blockDim.x + threadIdx.x;   // E*HID = 40.96M
    int e = idx >> 8, d = idx & 255;
    float al = attn[(size_t)e * NLAYER + layer];
    atomicAdd(&rst[(size_t)dst[e] * HID + d], al * fs[(size_t)src[e] * HID + d]);
}

// ---------------------------------------------------------------------------
// BatchNorm (training batch stats, biased var): one block per column.
// ---------------------------------------------------------------------------
__global__ __launch_bounds__(256)
void bn_stats_kernel(const float* __restrict__ X, float* __restrict__ mean,
                     float* __restrict__ var, int rows, int cols)
{
    int c = blockIdx.x;
    float s = 0.0f, s2 = 0.0f;
    for (int r = threadIdx.x; r < rows; r += 256) {
        float v = X[(size_t)r * cols + c];
        s += v; s2 += v * v;
    }
    __shared__ float sh[256], sh2[256];
    sh[threadIdx.x] = s; sh2[threadIdx.x] = s2;
    __syncthreads();
    for (int st = 128; st > 0; st >>= 1) {
        if (threadIdx.x < st) {
            sh[threadIdx.x]  += sh[threadIdx.x + st];
            sh2[threadIdx.x] += sh2[threadIdx.x + st];
        }
        __syncthreads();
    }
    if (threadIdx.x == 0) {
        float mu = sh[0] / rows;
        mean[c] = mu;
        var[c] = sh2[0] / rows - mu * mu;
    }
}

// BN + ReLU + graph-mean pooling into concat buffer (col offset layer*HID)
__global__ void bn_relu_pool_kernel(const float* __restrict__ X,
                                    const float* __restrict__ mean, const float* __restrict__ var,
                                    const float* __restrict__ g, const float* __restrict__ b,
                                    float* __restrict__ H, float* __restrict__ pooled, int layer)
{
    int idx = blockIdx.x * blockDim.x + threadIdx.x;   // NNODES*HID
    int d = idx & 255, n = idx >> 8;
    float v = g[d] * (X[idx] - mean[d]) * rsqrtf(var[d] + EPS_BN) + b[d];
    v = fmaxf(v, 0.0f);
    H[idx] = v;
    atomicAdd(&pooled[(size_t)(n / NPG) * (NLAYER * HID) + layer * HID + d],
              v * (1.0f / NPG));
}

__global__ void bn_relu_kernel(const float* __restrict__ X,
                               const float* __restrict__ mean, const float* __restrict__ var,
                               const float* __restrict__ g, const float* __restrict__ b,
                               float* __restrict__ out, int total)
{
    int idx = blockIdx.x * blockDim.x + threadIdx.x;
    if (idx >= total) return;
    int d = idx & 255;
    float v = g[d] * (X[idx] - mean[d]) * rsqrtf(var[d] + EPS_BN) + b[d];
    out[idx] = fmaxf(v, 0.0f);
}

// fc2: [200,256] @ [256,5] + b  (too skinny for WMMA)
__global__ void fc2_kernel(const float* __restrict__ X, const float* __restrict__ W,
                           const float* __restrict__ b, float* __restrict__ out, int total)
{
    int idx = blockIdx.x * blockDim.x + threadIdx.x;
    if (idx >= total) return;
    int row = idx / 5, o = idx % 5;
    float s = b[o];
    const float* xr = X + (size_t)row * HID;
    for (int k = 0; k < HID; ++k) s += xr[k] * W[k * 5 + o];
    out[idx] = s;
}

// ---------------------------------------------------------------------------
extern "C" void kernel_launch(void* const* d_in, const int* in_sizes, int n_in,
                              void* d_out, int out_size, void* d_ws, size_t ws_size,
                              hipStream_t stream)
{
    const float* x     = (const float*)d_in[0];
    const float* Ws0   = (const float*)d_in[1];
    const float* bs0   = (const float*)d_in[2];
    const float* Wd0   = (const float*)d_in[3];
    const float* bd0   = (const float*)d_in[4];
    const float* a0    = (const float*)d_in[5];
    const float* Wres0 = (const float*)d_in[6];
    const float* g0    = (const float*)d_in[7];
    const float* be0   = (const float*)d_in[8];
    const float* Ws_r  = (const float*)d_in[9];
    const float* bs_r  = (const float*)d_in[10];
    const float* Wd_r  = (const float*)d_in[11];
    const float* bd_r  = (const float*)d_in[12];
    const float* a_r   = (const float*)d_in[13];
    const float* g_r   = (const float*)d_in[14];
    const float* be_r  = (const float*)d_in[15];
    const float* fc1_W = (const float*)d_in[16];
    const float* fc1_b = (const float*)d_in[17];
    const float* gfc   = (const float*)d_in[18];
    const float* bfc   = (const float*)d_in[19];
    const float* fc2_W = (const float*)d_in[20];
    const float* fc2_b = (const float*)d_in[21];
    const int*   src   = (const int*)d_in[22];
    const int*   dst   = (const int*)d_in[23];

    // Workspace partition (floats, 256B-aligned chunks)
    float* ws = (float*)d_ws;
    size_t off = 0;
    auto alloc = [&](size_t n) { float* p = ws + off; off += (n + 63) & ~(size_t)63; return p; };
    float*    h      = alloc((size_t)NNODES * HID);
    float*    fs     = alloc((size_t)NNODES * HID);
    float*    fd     = alloc((size_t)NNODES * HID);
    float*    rst    = alloc((size_t)NNODES * HID);
    float*    score  = alloc(NEDGES);
    float*    ex     = alloc(NEDGES);
    unsigned* nmax   = (unsigned*)alloc(NNODES);
    float*    den    = alloc(NNODES);
    float*    pooled = alloc((size_t)NGRAPH * NLAYER * HID);
    float*    t1     = alloc((size_t)NGRAPH * HID);
    float*    t2     = alloc((size_t)NGRAPH * HID);
    float*    bmean  = alloc(HID);
    float*    bvar   = alloc(HID);

    float* feat_out = (float*)d_out;                    // [200,5]
    float* attn_out = (float*)d_out + NGRAPH * 5;       // [E,4]

    const int NH = NNODES * HID;   // 5,120,000 (mult of 256)
    const int EH = NEDGES * HID;   // 40,960,000 (mult of 256)

    fill_f32<<<(NGRAPH * NLAYER * HID + 255) / 256, 256, 0, stream>>>(
        pooled, 0.0f, NGRAPH * NLAYER * HID);

    dim3 gemm_grid_n((NNODES + 63) / 64, HID / 64);

    for (int l = 0; l < NLAYER; ++l) {
        const float *act, *Ws, *bs, *Wd, *bd, *av, *g, *be;
        int K;
        if (l == 0) {
            act = x; K = IN_DIM;
            Ws = Ws0; bs = bs0; Wd = Wd0; bd = bd0; av = a0; g = g0; be = be0;
        } else {
            act = h; K = HID;
            int m = l - 1;
            Ws = Ws_r + (size_t)m * HID * HID; bs = bs_r + m * HID;
            Wd = Wd_r + (size_t)m * HID * HID; bd = bd_r + m * HID;
            av = a_r + m * HID; g = g_r + m * HID; be = be_r + m * HID;
        }

        gemm_wmma_kernel<<<gemm_grid_n, 128, 0, stream>>>(act, Ws, bs, fs, NNODES, K, HID);
        gemm_wmma_kernel<<<gemm_grid_n, 128, 0, stream>>>(act, Wd, bd, fd, NNODES, K, HID);

        // rst = residual (then scatter-add attention-weighted messages)
        if (l == 0)
            gemm_wmma_kernel<<<gemm_grid_n, 128, 0, stream>>>(x, Wres0, nullptr, rst, NNODES, IN_DIM, HID);
        else
            copy_f32<<<NH / 256, 256, 0, stream>>>(rst, h, NH);

        fill_u32<<<(NNODES + 255) / 256, 256, 0, stream>>>(nmax, 0u, NNODES);
        fill_f32<<<(NNODES + 255) / 256, 256, 0, stream>>>(den, 0.0f, NNODES);

        edge_score_kernel<<<(NEDGES + 7) / 8, 256, 0, stream>>>(fs, fd, av, src, dst, score, nmax, NEDGES);
        edge_exp_kernel<<<NEDGES / 256, 256, 0, stream>>>(score, dst, nmax, ex, den, NEDGES);
        alpha_kernel<<<NEDGES / 256, 256, 0, stream>>>(ex, den, dst, attn_out, l, NEDGES);
        agg_kernel<<<EH / 256, 256, 0, stream>>>(fs, attn_out, src, dst, rst, l);

        bn_stats_kernel<<<HID, 256, 0, stream>>>(rst, bmean, bvar, NNODES, HID);
        bn_relu_pool_kernel<<<NH / 256, 256, 0, stream>>>(rst, bmean, bvar, g, be, h, pooled, l);
    }

    // fc1: [200, 1024] @ [1024, 256] + b -> BN -> ReLU -> fc2
    dim3 gemm_grid_g((NGRAPH + 63) / 64, HID / 64);
    gemm_wmma_kernel<<<gemm_grid_g, 128, 0, stream>>>(pooled, fc1_W, fc1_b, t1,
                                                      NGRAPH, NLAYER * HID, HID);
    bn_stats_kernel<<<HID, 256, 0, stream>>>(t1, bmean, bvar, NGRAPH, HID);
    bn_relu_kernel<<<(NGRAPH * HID + 255) / 256, 256, 0, stream>>>(t1, bmean, bvar, gfc, bfc,
                                                                   t2, NGRAPH * HID);
    fc2_kernel<<<(NGRAPH * 5 + 255) / 256, 256, 0, stream>>>(t2, fc2_W, fc2_b,
                                                             feat_out, NGRAPH * 5);
}
